// GCN_27023934226530
// MI455X (gfx1250) — matrix operand order repaired and checked by reading, not compiled, not run on hardware
//
#include <hip/hip_runtime.h>
#include <hip/hip_bf16.h>

// ---------------------------------------------------------------------------
// Types for CDNA5 WMMA (wave32, 16x16x32 bf16 -> f32)
// ---------------------------------------------------------------------------
typedef __attribute__((ext_vector_type(16))) __bf16 v16bf;
typedef __attribute__((ext_vector_type(8)))  float  v8f;

union AB16 { uint4 u[2]; v16bf v; };

__device__ __forceinline__ unsigned short f2bf(float f) {
    // round-to-nearest-even float -> bf16 (raw bits)
    unsigned int u = __float_as_uint(f);
    unsigned int r = u + 0x7FFFu + ((u >> 16) & 1u);
    return (unsigned short)(r >> 16);
}

// ---------------------------------------------------------------------------
// Graph preprocessing
// ---------------------------------------------------------------------------
__global__ void fill_f32(float* p, float v, size_t n) {
    size_t i = (size_t)blockIdx.x * blockDim.x + threadIdx.x;
    if (i < n) p[i] = v;
}

__global__ void deg_edges(float* __restrict__ deg, const int* __restrict__ ei, int E) {
    int e = blockIdx.x * blockDim.x + threadIdx.x;
    if (e < E) atomicAdd(&deg[ei[(size_t)E + e]], 1.0f);
}

__global__ void rsqrt_k(const float* __restrict__ deg, float* __restrict__ dinv, int N) {
    int i = blockIdx.x * blockDim.x + threadIdx.x;
    if (i < N) dinv[i] = rsqrtf(deg[i]);
}

__global__ void norm_k(const int* __restrict__ ei, const float* __restrict__ dinv,
                       float* __restrict__ nrm, int E, int N) {
    int e = blockIdx.x * blockDim.x + threadIdx.x;
    if (e >= E + N) return;
    if (e < E) {
        nrm[e] = dinv[ei[e]] * dinv[ei[(size_t)E + e]];
    } else {
        float d = dinv[e - E];
        nrm[e] = d * d;
    }
}

// ---------------------------------------------------------------------------
// BatchNorm (training mode, biased variance) over node axis
// ---------------------------------------------------------------------------
__global__ void bn_stats(const float* __restrict__ x, float* __restrict__ s1,
                         float* __restrict__ s2, int N, int F, int rowsPerBlock) {
    int r0 = blockIdx.x * rowsPerBlock;
    int r1 = r0 + rowsPerBlock; if (r1 > N) r1 = N;
    int c0 = threadIdx.x;
    int c1 = threadIdx.x + blockDim.x;          // F <= 2*blockDim assumed (F<=512)
    float a0 = 0.f, q0 = 0.f, a1 = 0.f, q1 = 0.f;
    for (int r = r0; r < r1; ++r) {
        const float* row = x + (size_t)r * F;
        if (c0 < F) { float v = row[c0]; a0 += v; q0 += v * v; }
        if (c1 < F) { float w = row[c1]; a1 += w; q1 += w * w; }
    }
    if (c0 < F) { atomicAdd(&s1[c0], a0); atomicAdd(&s2[c0], q0); }
    if (c1 < F) { atomicAdd(&s1[c1], a1); atomicAdd(&s2[c1], q1); }
}

__global__ void bn_coef(const float* __restrict__ s1, const float* __restrict__ s2,
                        const float* __restrict__ gamma, const float* __restrict__ beta,
                        float* __restrict__ scale, float* __restrict__ shift, int N, int F) {
    int c = blockIdx.x * blockDim.x + threadIdx.x;
    if (c >= F) return;
    float inv = 1.0f / (float)N;
    float mean = s1[c] * inv;
    float var  = fmaxf(s2[c] * inv - mean * mean, 0.0f);
    float sc   = gamma[c] * rsqrtf(var + 1e-5f);
    scale[c] = sc;
    shift[c] = beta[c] - mean * sc;
}

__global__ void bn_apply_bf16(const float* __restrict__ x, const float* __restrict__ scale,
                              const float* __restrict__ shift, unsigned short* __restrict__ out,
                              size_t total, int F) {
    size_t i = (size_t)blockIdx.x * blockDim.x + threadIdx.x;
    if (i >= total) return;
    int c = (int)(i % (size_t)F);
    out[i] = f2bf(fmaf(x[i], scale[c], shift[c]));
}

__global__ void bn_apply_f32(const float* __restrict__ x, const float* __restrict__ scale,
                             const float* __restrict__ shift, float* __restrict__ out,
                             size_t total, int F) {
    size_t i = (size_t)blockIdx.x * blockDim.x + threadIdx.x;
    if (i >= total) return;
    int c = (int)(i % (size_t)F);
    out[i] = fmaf(x[i], scale[c], shift[c]);
}

// ---------------------------------------------------------------------------
// Weight cast + transpose: W[K,Ncols] f32  ->  Wt[Ncols,K] bf16
// ---------------------------------------------------------------------------
__global__ void wcast_t(const float* __restrict__ W, unsigned short* __restrict__ Wt,
                        int K, int Ncols) {
    int i = blockIdx.x * blockDim.x + threadIdx.x;
    if (i >= K * Ncols) return;
    int k = i / Ncols;
    int n = i - k * Ncols;
    Wt[(size_t)n * K + k] = f2bf(W[i]);
}

// ---------------------------------------------------------------------------
// WMMA GEMM with LDS-staged B: C[N,Ncols](f32) = A[N,K](bf16) * Bt[Ncols,K]^T
// Block = 8 waves = 128 rows x 64 cols; the 64-col x 32-K B chunk is loaded
// into LDS once per K-step and shared by all 8 waves (8x less B traffic).
// Fragment layouts used here (A/B in contiguous K order, not the ISA table order):
//   A 16x32: row=lane&15; lanes<16 hold K 0..15, lanes>=16 hold K 16..31
//   B 32x16: col=lane&15; lanes<16 hold K 0..15, lanes>=16 hold K 16..31
//   C 16x16: VGPR v -> row v + 8*(lane>=16), col = lane&15
// LDS column stride = 40 shorts (80B): ds_load_b128 lane addresses land on
// 16 distinct bank groups (20*i mod 64 is a permutation for i=0..15).
// ---------------------------------------------------------------------------
#define LDS_STRIDE 40

__global__ __launch_bounds__(256)
void gemm_bf16_wmma(const unsigned short* __restrict__ A,
                    const unsigned short* __restrict__ Bt,
                    float* __restrict__ C, int N, int K, int Ncols) {
    __shared__ unsigned short lb[64 * LDS_STRIDE];   // 5 KB

    const int tid  = threadIdx.x;
    const int lane = tid & 31;
    const int wav  = tid >> 5;
    const int nBlkN = Ncols >> 6;                    // 64-wide column tiles
    const int blkM  = blockIdx.x / nBlkN;
    const int blkN  = blockIdx.x - blkM * nBlkN;
    const int col0  = blkN * 64;

    const int hh  = lane >> 4;                       // half-wave select
    const int l15 = lane & 15;

    int arow = blkM * 128 + wav * 16 + l15;
    if (arow >= N) arow = N - 1;                     // clamp (stores guarded)
    const unsigned short* ap = A + (size_t)arow * K + (hh ? 16 : 0);

    // global->LDS cooperative fill mapping: 256 threads x 16B = 64x32 bf16 chunk
    const int fc = tid >> 2;                         // column 0..63
    const int fp = (tid & 3) * 8;                    // K-part 0,8,16,24
    const unsigned short* bsrc = Bt + (size_t)(col0 + fc) * K + fp;
    unsigned short*       bdst = &lb[fc * LDS_STRIDE + fp];

    // per-lane LDS fragment base: col = s*16 + l15, K-half = hh*16
    const unsigned short* lbase = &lb[l15 * LDS_STRIDE + hh * 16];

    v8f acc0 = {}, acc1 = {}, acc2 = {}, acc3 = {};

    for (int k0 = 0; k0 < K; k0 += 32) {
        uint4 bw = *(const uint4*)(bsrc + k0);       // overlaps prev step's WMMAs
        __syncthreads();                             // prev reads done
        *(uint4*)bdst = bw;
        __syncthreads();                             // chunk visible

        AB16 a;
        a.u[0] = *(const uint4*)(ap + k0);           // K base..base+7
        a.u[1] = *(const uint4*)(ap + k0 + 8);       // K base+8..base+15
        if (k0 + 32 < K) __builtin_prefetch(ap + k0 + 32, 0, 1);  // global_prefetch_b8

        AB16 b0, b1, b2, b3;
        b0.u[0] = *(const uint4*)(lbase);
        b0.u[1] = *(const uint4*)(lbase + 8);
        b1.u[0] = *(const uint4*)(lbase + 16 * LDS_STRIDE);
        b1.u[1] = *(const uint4*)(lbase + 16 * LDS_STRIDE + 8);
        b2.u[0] = *(const uint4*)(lbase + 32 * LDS_STRIDE);
        b2.u[1] = *(const uint4*)(lbase + 32 * LDS_STRIDE + 8);
        b3.u[0] = *(const uint4*)(lbase + 48 * LDS_STRIDE);
        b3.u[1] = *(const uint4*)(lbase + 48 * LDS_STRIDE + 8);

        acc0 = __builtin_amdgcn_wmma_f32_16x16x32_bf16(false, a.v, false, b0.v, (short)0, acc0, false, false);
        acc1 = __builtin_amdgcn_wmma_f32_16x16x32_bf16(false, a.v, false, b1.v, (short)0, acc1, false, false);
        acc2 = __builtin_amdgcn_wmma_f32_16x16x32_bf16(false, a.v, false, b2.v, (short)0, acc2, false, false);
        acc3 = __builtin_amdgcn_wmma_f32_16x16x32_bf16(false, a.v, false, b3.v, (short)0, acc3, false, false);
    }

    const int crow0 = blkM * 128 + wav * 16 + (hh ? 8 : 0);
    float* cp = C + (size_t)crow0 * Ncols + col0 + l15;
#pragma unroll
    for (int v = 0; v < 8; ++v) {
        if (crow0 + v < N) {
            float* q = cp + (size_t)v * Ncols;
            q[0]  = acc0[v];
            q[16] = acc1[v];
            q[32] = acc2[v];
            q[48] = acc3[v];
        }
    }
}

// ---------------------------------------------------------------------------
// SpMM scatter: O[dst] += norm * T[src]  (64 threads per edge, float4 chunks)
// ---------------------------------------------------------------------------
__global__ void spmm_scatter(const float* __restrict__ T, float* __restrict__ O,
                             const int* __restrict__ ei, const float* __restrict__ nrm,
                             int E, int N, int Hdim) {
    unsigned long long t = (unsigned long long)blockIdx.x * blockDim.x + threadIdx.x;
    int per = Hdim >> 2;
    unsigned long long e = t / (unsigned long long)per;
    int f = (int)(t - e * (unsigned long long)per) << 2;
    unsigned long long total = (unsigned long long)E + (unsigned long long)N;
    if (e >= total) return;
    int s, d;
    if (e < (unsigned long long)E) {
        s = ei[e];
        d = ei[(size_t)E + e];
    } else {
        s = d = (int)(e - (unsigned long long)E);
    }
    float w = nrm[e];
    float4 v = *(const float4*)(T + (size_t)s * Hdim + f);
    float* o = O + (size_t)d * Hdim + f;
    atomicAdd(o + 0, w * v.x);
    atomicAdd(o + 1, w * v.y);
    atomicAdd(o + 2, w * v.z);
    atomicAdd(o + 3, w * v.w);
}

// ---------------------------------------------------------------------------
// Bias add (+ optional bf16 cast for the next layer input)
// ---------------------------------------------------------------------------
__global__ void bias_cast_bf16(const float* __restrict__ O, const float* __restrict__ b,
                               unsigned short* __restrict__ out, size_t total, int Hdim) {
    size_t i = (size_t)blockIdx.x * blockDim.x + threadIdx.x;
    if (i >= total) return;
    out[i] = f2bf(O[i] + b[i % (size_t)Hdim]);
}

__global__ void bias_f32(const float* __restrict__ O, const float* __restrict__ b,
                         float* __restrict__ out, size_t total, int Hdim) {
    size_t i = (size_t)blockIdx.x * blockDim.x + threadIdx.x;
    if (i >= total) return;
    out[i] = O[i] + b[i % (size_t)Hdim];
}

// ---------------------------------------------------------------------------
// Final dense [N,H] @ [H,5] + bias, sigmoid
// ---------------------------------------------------------------------------
__global__ void dense_sigmoid(const float* __restrict__ hmat, const float* __restrict__ Wd,
                              const float* __restrict__ bd, float* __restrict__ out,
                              int N, int Hdim, int Cout) {
    int r = blockIdx.x * blockDim.x + threadIdx.x;
    if (r >= N) return;
    float acc[8];
    for (int j = 0; j < Cout; ++j) acc[j] = bd[j];
    const float* hr = hmat + (size_t)r * Hdim;
    for (int k = 0; k < Hdim; k += 4) {
        float4 h4 = *(const float4*)(hr + k);
        const float* w = Wd + (size_t)k * Cout;
        for (int j = 0; j < Cout; ++j)
            acc[j] += h4.x * w[j] + h4.y * w[Cout + j] + h4.z * w[2 * Cout + j] + h4.w * w[3 * Cout + j];
    }
    for (int j = 0; j < Cout; ++j)
        out[(size_t)r * Cout + j] = 1.0f / (1.0f + __expf(-acc[j]));
}

// ---------------------------------------------------------------------------
// Host orchestration
// ---------------------------------------------------------------------------
static inline int ceil_div(long long a, long long b) { return (int)((a + b - 1) / b); }

extern "C" void kernel_launch(void* const* d_in, const int* in_sizes, int n_in,
                              void* d_out, int out_size, void* d_ws, size_t ws_size,
                              hipStream_t stream) {
    const float* x      = (const float*)d_in[0];
    const int*   ei     = (const int*)d_in[1];
    const float* gamma1 = (const float*)d_in[2];
    const float* beta1  = (const float*)d_in[3];
    const float* W1 = (const float*)d_in[4];
    const float* b1 = (const float*)d_in[5];
    const float* W2 = (const float*)d_in[6];
    const float* b2 = (const float*)d_in[7];
    const float* W3 = (const float*)d_in[8];
    const float* b3 = (const float*)d_in[9];
    const float* W4 = (const float*)d_in[10];
    const float* b4 = (const float*)d_in[11];
    const float* gamma2 = (const float*)d_in[12];
    const float* beta2  = (const float*)d_in[13];
    const float* Wd = (const float*)d_in[14];
    const float* bd = (const float*)d_in[15];

    const int F = in_sizes[2];            // 512
    const int H = in_sizes[12];           // 256
    const int N = in_sizes[0] / F;        // 50000
    const int E = in_sizes[1] / 2;        // 800000
    const int Cout = in_sizes[15];        // 5

    // --- carve workspace (all offsets 256B aligned) ---
    char* ws = (char*)d_ws;
    size_t off = 0;
    auto take = [&](size_t bytes) -> char* {
        char* p = ws + off;
        off = (off + bytes + 255) & ~(size_t)255;
        return p;
    };
    unsigned short* Xb  = (unsigned short*)take((size_t)N * F * 2);   // bf16 activations
    float*          T   = (float*)take((size_t)N * H * 4);            // GEMM output
    float*          O   = (float*)take((size_t)N * H * 4);            // SpMM output
    unsigned short* Wt1 = (unsigned short*)take((size_t)F * H * 2);
    unsigned short* Wt2 = (unsigned short*)take((size_t)H * H * 2);
    unsigned short* Wt3 = (unsigned short*)take((size_t)H * H * 2);
    unsigned short* Wt4 = (unsigned short*)take((size_t)H * H * 2);
    float* deg  = (float*)take((size_t)N * 4);
    float* dinv = (float*)take((size_t)N * 4);
    float* nrm  = (float*)take((size_t)(E + N) * 4);
    float* s1   = (float*)take((size_t)F * 4);
    float* s2   = (float*)take((size_t)F * 4);
    float* scl  = (float*)take((size_t)F * 4);
    float* shf  = (float*)take((size_t)F * 4);

    const int TB = 256;

    // --- graph norms ---
    fill_f32<<<ceil_div(N, TB), TB, 0, stream>>>(deg, 1.0f, (size_t)N);  // self loops
    deg_edges<<<ceil_div(E, TB), TB, 0, stream>>>(deg, ei, E);
    rsqrt_k<<<ceil_div(N, TB), TB, 0, stream>>>(deg, dinv, N);
    norm_k<<<ceil_div((long long)E + N, TB), TB, 0, stream>>>(ei, dinv, nrm, E, N);

    // --- BN1 + cast to bf16 ---
    fill_f32<<<ceil_div(2 * F, TB), TB, 0, stream>>>(s1, 0.0f, (size_t)2 * F); // s1,s2 contiguous
    bn_stats<<<ceil_div(N, 64), TB, 0, stream>>>(x, s1, s2, N, F, 64);
    bn_coef<<<ceil_div(F, TB), TB, 0, stream>>>(s1, s2, gamma1, beta1, scl, shf, N, F);
    bn_apply_bf16<<<ceil_div((long long)N * F, TB), TB, 0, stream>>>(x, scl, shf, Xb, (size_t)N * F, F);

    // --- weight prep (cast + transpose) ---
    wcast_t<<<ceil_div((long long)F * H, TB), TB, 0, stream>>>(W1, Wt1, F, H);
    wcast_t<<<ceil_div((long long)H * H, TB), TB, 0, stream>>>(W2, Wt2, H, H);
    wcast_t<<<ceil_div((long long)H * H, TB), TB, 0, stream>>>(W3, Wt3, H, H);
    wcast_t<<<ceil_div((long long)H * H, TB), TB, 0, stream>>>(W4, Wt4, H, H);

    const int gemmBlocks = ceil_div(N, 128) * (H / 64);
    const long long spmmThreads = ((long long)E + N) * (H / 4);
    const size_t nh = (size_t)N * H;

    // --- layer 1: GEMM(Xb[F], Wt1) -> T ; scatter -> O ; +b1 -> Xb (bf16) ---
    gemm_bf16_wmma<<<gemmBlocks, TB, 0, stream>>>(Xb, Wt1, T, N, F, H);
    fill_f32<<<ceil_div((long long)nh, TB), TB, 0, stream>>>(O, 0.0f, nh);
    spmm_scatter<<<ceil_div(spmmThreads, TB), TB, 0, stream>>>(T, O, ei, nrm, E, N, H);
    bias_cast_bf16<<<ceil_div((long long)nh, TB), TB, 0, stream>>>(O, b1, Xb, nh, H);

    // --- layers 2,3 ---
    gemm_bf16_wmma<<<gemmBlocks, TB, 0, stream>>>(Xb, Wt2, T, N, H, H);
    fill_f32<<<ceil_div((long long)nh, TB), TB, 0, stream>>>(O, 0.0f, nh);
    spmm_scatter<<<ceil_div(spmmThreads, TB), TB, 0, stream>>>(T, O, ei, nrm, E, N, H);
    bias_cast_bf16<<<ceil_div((long long)nh, TB), TB, 0, stream>>>(O, b2, Xb, nh, H);

    gemm_bf16_wmma<<<gemmBlocks, TB, 0, stream>>>(Xb, Wt3, T, N, H, H);
    fill_f32<<<ceil_div((long long)nh, TB), TB, 0, stream>>>(O, 0.0f, nh);
    spmm_scatter<<<ceil_div(spmmThreads, TB), TB, 0, stream>>>(T, O, ei, nrm, E, N, H);
    bias_cast_bf16<<<ceil_div((long long)nh, TB), TB, 0, stream>>>(O, b3, Xb, nh, H);

    // --- layer 4: keep fp32 for BN2 ---
    gemm_bf16_wmma<<<gemmBlocks, TB, 0, stream>>>(Xb, Wt4, T, N, H, H);
    fill_f32<<<ceil_div((long long)nh, TB), TB, 0, stream>>>(O, 0.0f, nh);
    spmm_scatter<<<ceil_div(spmmThreads, TB), TB, 0, stream>>>(T, O, ei, nrm, E, N, H);
    bias_f32<<<ceil_div((long long)nh, TB), TB, 0, stream>>>(O, b4, T, nh, H);   // h4 -> T (f32)

    // --- BN2 (fp32) -> O ---
    fill_f32<<<ceil_div(2 * H, TB), TB, 0, stream>>>(s1, 0.0f, (size_t)2 * H);
    bn_stats<<<ceil_div(N, 64), TB, 0, stream>>>(T, s1, s2, N, H, 64);
    bn_coef<<<ceil_div(H, TB), TB, 0, stream>>>(s1, s2, gamma2, beta2, scl, shf, N, H);
    bn_apply_f32<<<ceil_div((long long)nh, TB), TB, 0, stream>>>(T, scl, shf, O, nh, H);

    // --- final dense + sigmoid -> d_out ---
    dense_sigmoid<<<ceil_div(N, TB), TB, 0, stream>>>(O, Wd, bd, (float*)d_out, N, H, Cout);
}
